// UnifiedMoE_23587960389767
// MI455X (gfx1250) — matrix operand order repaired and checked by reading, not compiled, not run on hardware
//
#include <hip/hip_runtime.h>
#include <hip/hip_bf16.h>
#include <stdint.h>

#define DIM   1024
#define E_    8
#define HID   4096
#define NTOK  2048
#define MT    32            // tokens per ffn block
#define CH    256           // hid chunk per iteration

#define XS_STRIDE 1064      // bf16 per Xs row (2128B: 16B-aligned, 20-bank shift)
#define HS_STRIDE 296       // bf16 per hs row (592B: 16B-aligned, 20-bank shift)
#define SMEM_BYTES ((MT * XS_STRIDE + MT * HS_STRIDE) * 2 + MT * 4 * 3)

typedef __attribute__((ext_vector_type(16))) __bf16          v16bf;
typedef __attribute__((ext_vector_type(8)))  __bf16          v8bf;
typedef __attribute__((ext_vector_type(8)))  float           v8f;

union AFrag { v16bf v; v8bf h[2]; };
union BFrag { v16bf v; __bf16 e[16]; };

// tanh-approx gelu via sigmoid identity: 0.5*v*(1+tanh(u)) = v * rcp(1+exp2(-2*log2e*u))
// branch-free: v_exp_f32 + v_rcp_f32 (TRANS pipe, co-executes with WMMA)
__device__ __forceinline__ float gelu_tanh(float v) {
  float u = v * 0.7978845608028654f * (1.f + 0.044715f * v * v);
  float s = __builtin_amdgcn_rcpf(1.f + __builtin_amdgcn_exp2f(-2.8853900817779268f * u));
  return v * s;
}

// ---------------------------------------------------------------- init ----
__global__ void moe_init(int* cnt) {
  if (threadIdx.x < E_) cnt[threadIdx.x] = 0;
}

// -------------------------------------------------------------- router ----
__global__ __launch_bounds__(256)
void moe_router(const float* __restrict__ x, const float* __restrict__ Wr,
                const float* __restrict__ br, int* __restrict__ cnt,
                int* __restrict__ list, float* __restrict__ wlist,
                float* __restrict__ probs_out, float* __restrict__ zsq) {
  int wave = threadIdx.x >> 5, lane = threadIdx.x & 31;
  int t = blockIdx.x * 8 + wave;

  float acc[E_];
#pragma unroll
  for (int e = 0; e < E_; ++e) acc[e] = 0.f;

  const float* xr = x + (size_t)t * DIM;
  for (int d = lane; d < DIM; d += 32) {
    float xv = xr[d];
    const float* wr = Wr + (size_t)d * E_;
#pragma unroll
    for (int e = 0; e < E_; ++e) acc[e] = fmaf(xv, wr[e], acc[e]);
  }
#pragma unroll
  for (int off = 16; off >= 1; off >>= 1) {
#pragma unroll
    for (int e = 0; e < E_; ++e) acc[e] += __shfl_xor(acc[e], off, 32);
  }

  if (lane == 0) {
    float lg[E_], mx = -1e30f;
#pragma unroll
    for (int e = 0; e < E_; ++e) { lg[e] = acc[e] + br[e]; mx = fmaxf(mx, lg[e]); }
    float p[E_], s = 0.f;
#pragma unroll
    for (int e = 0; e < E_; ++e) { p[e] = expf(lg[e] - mx); s += p[e]; }
    float lse = logf(s) + mx;
    zsq[t] = lse * lse;
    float inv = 1.f / s;
#pragma unroll
    for (int e = 0; e < E_; ++e) { p[e] *= inv; probs_out[t * E_ + e] = p[e]; }
    int i0 = 0;
#pragma unroll
    for (int e = 1; e < E_; ++e) if (p[e] > p[i0]) i0 = e;
    int i1 = (i0 == 0) ? 1 : 0;
#pragma unroll
    for (int e = 0; e < E_; ++e) if (e != i0 && p[e] > p[i1]) i1 = e;
    float s2 = p[i0] + p[i1] + 1e-12f;
    int pos0 = atomicAdd(&cnt[i0], 1);
    list[i0 * NTOK + pos0] = t * 2;     wlist[i0 * NTOK + pos0] = p[i0] / s2;
    int pos1 = atomicAdd(&cnt[i1], 1);
    list[i1 * NTOK + pos1] = t * 2 + 1; wlist[i1 * NTOK + pos1] = p[i1] / s2;
  }
}

// ----------------------------------------------------------------- ffn ----
// Block = (expert e, tile of 32 tokens), 512 threads = 16 waves.
// GEMM1: wave w computes h columns [hc*256 + w*16, +16) for both 16-row tiles.
// GEMM2: wave w accumulates out columns [w*64, +64) for both tiles.
// Every B-fragment (global fp32 -> bf16 via v_cvt_pk_bf16_f32) feeds 2 WMMAs.
__global__ __launch_bounds__(512)
void moe_ffn(const float* __restrict__ x,  const float* __restrict__ W1,
             const float* __restrict__ b1, const float* __restrict__ W2,
             const float* __restrict__ b2, const int* __restrict__ cnt,
             const int* __restrict__ list, const float* __restrict__ wlist,
             float* __restrict__ Y) {
  int e    = blockIdx.x >> 6;            // 64 tiles / expert
  int tile = blockIdx.x & 63;
  int n_e  = cnt[e];
  if (tile * MT >= n_e) return;          // uniform per block

  extern __shared__ __align__(16) __bf16 smem[];
  __bf16 (*Xs)[XS_STRIDE] = (__bf16 (*)[XS_STRIDE])smem;
  __bf16 (*hs)[HS_STRIDE] = (__bf16 (*)[HS_STRIDE])(smem + MT * XS_STRIDE);
  int*   s_row = (int*)(smem + MT * XS_STRIDE + MT * HS_STRIDE);
  int*   s_tok = s_row + MT;
  float* s_w   = (float*)(s_tok + MT);

  int tid = threadIdx.x;
  if (tid < MT) {
    int idx = tile * MT + tid;
    if (idx < n_e) {
      int a = list[e * NTOK + idx];
      s_row[tid] = a;                    // Y row = token*2 + slot
      s_tok[tid] = a >> 1;
      s_w[tid]   = wlist[e * NTOK + idx];
    } else {
      s_row[tid] = 2 * NTOK;             // dummy row (never read back)
      s_tok[tid] = -1;
      s_w[tid]   = 0.f;
    }
  }
  __syncthreads();

  // Stage X tile as bf16 (float4 loads + native converts).
  for (int i = tid; i < MT * (DIM / 4); i += 512) {
    int m  = i >> 8;
    int d4 = (i & 255) * 4;
    int tok = s_tok[m];
    float4 v = (tok >= 0) ? *(const float4*)(x + (size_t)tok * DIM + d4)
                          : make_float4(0.f, 0.f, 0.f, 0.f);
    Xs[m][d4 + 0] = (__bf16)v.x;
    Xs[m][d4 + 1] = (__bf16)v.y;
    Xs[m][d4 + 2] = (__bf16)v.z;
    Xs[m][d4 + 3] = (__bf16)v.w;
  }
  __syncthreads();

  int wv    = tid >> 5;                  // 0..15
  int lane  = tid & 31;
  int ln    = lane & 15;
  int hi    = lane >> 4;
  int abase = hi * 16;                   // A-frag K base
  int kboff = hi * 16;                   // B-frag K base

  const float* W1e = W1 + (size_t)e * DIM * HID;
  const float* W2e = W2 + (size_t)e * HID * DIM;

  v8f zero = {};
  v8f accA[4], accB[4];
#pragma unroll
  for (int i = 0; i < 4; ++i) { accA[i] = zero; accB[i] = zero; }

  for (int hc = 0; hc < HID / CH; ++hc) {
    // ---- GEMM1: h[32 x 16 cols of this wave] = X[32,1024] @ W1 ----
    v8f c1a = zero, c1b = zero;
    int ncol = hc * CH + wv * 16 + ln;
    const float* bp1 = W1e + ncol;
#pragma unroll 2
    for (int ks = 0; ks < DIM / 32; ++ks) {
      int kb = ks * 32;
      AFrag a0, a1;
      a0.h[0] = *(const v8bf*)&Xs[ln][kb + abase];
      a0.h[1] = *(const v8bf*)&Xs[ln][kb + 8 + abase];
      a1.h[0] = *(const v8bf*)&Xs[16 + ln][kb + abase];
      a1.h[1] = *(const v8bf*)&Xs[16 + ln][kb + 8 + abase];
      BFrag b;
      const float* p = bp1 + (size_t)(kb + kboff) * HID;
#pragma unroll
      for (int i = 0; i < 16; ++i) b.e[i] = (__bf16)p[(size_t)i * HID];
      c1a = __builtin_amdgcn_wmma_f32_16x16x32_bf16(
                false, a0.v, false, b.v, (short)0, c1a, false, false);
      c1b = __builtin_amdgcn_wmma_f32_16x16x32_bf16(
                false, a1.v, false, b.v, (short)0, c1b, false, false);
    }

    __syncthreads();                     // previous GEMM2 done reading hs
    float bias1 = b1[e * HID + ncol];
    int hcol = wv * 16 + ln;
#pragma unroll
    for (int r = 0; r < 8; ++r) {
      hs[r + hi * 8][hcol]      = (__bf16)gelu_tanh(c1a[r] + bias1);
      hs[16 + r + hi * 8][hcol] = (__bf16)gelu_tanh(c1b[r] + bias1);
    }
    __syncthreads();                     // hs chunk ready

    // ---- GEMM2: acc[32 x 64 cols of this wave] += h[32,256] @ W2 ----
#pragma unroll 2
    for (int ks = 0; ks < CH / 32; ++ks) {
      int kb = ks * 32;
      AFrag a0, a1;
      a0.h[0] = *(const v8bf*)&hs[ln][kb + abase];
      a0.h[1] = *(const v8bf*)&hs[ln][kb + 8 + abase];
      a1.h[0] = *(const v8bf*)&hs[16 + ln][kb + abase];
      a1.h[1] = *(const v8bf*)&hs[16 + ln][kb + 8 + abase];
      const float* prow = W2e + (size_t)(hc * CH + kb + kboff) * DIM
                        + wv * 64 + ln;
#pragma unroll
      for (int nt = 0; nt < 4; ++nt) {
        BFrag b;
        const float* p = prow + nt * 16;
#pragma unroll
        for (int i = 0; i < 16; ++i) b.e[i] = (__bf16)p[(size_t)i * DIM];
        accA[nt] = __builtin_amdgcn_wmma_f32_16x16x32_bf16(
                       false, a0.v, false, b.v, (short)0, accA[nt], false, false);
        accB[nt] = __builtin_amdgcn_wmma_f32_16x16x32_bf16(
                       false, a1.v, false, b.v, (short)0, accB[nt], false, false);
      }
    }
  }

  // ---- epilogue: bias2, per-token combine weight, scatter to Y rows ----
#pragma unroll
  for (int nt = 0; nt < 4; ++nt) {
    int n = wv * 64 + nt * 16 + ln;
    float bias2 = b2[e * DIM + n];
#pragma unroll
    for (int r = 0; r < 8; ++r) {
      int m0 = r + hi * 8;
      int m1 = 16 + m0;
      Y[(size_t)s_row[m0] * DIM + n] = (accA[nt][r] + bias2) * s_w[m0];
      Y[(size_t)s_row[m1] * DIM + n] = (accB[nt][r] + bias2) * s_w[m1];
    }
  }
}

// ------------------------------------------------------------- combine ----
__global__ __launch_bounds__(256)
void moe_combine(const float* __restrict__ Y, float* __restrict__ out) {
  int i  = blockIdx.x * 256 + threadIdx.x;      // float4 index
  int t  = i >> 8;                              // 256 float4 per row
  int d4 = i & 255;
  const float4* ya = (const float4*)(Y + (size_t)(2 * t) * DIM) + d4;
  const float4* yb = (const float4*)(Y + (size_t)(2 * t + 1) * DIM) + d4;
  float4 a = *ya, b = *yb;
  ((float4*)out)[i] = make_float4(a.x + b.x, a.y + b.y, a.z + b.z, a.w + b.w);
}

// --------------------------------------------------------------- stats ----
__global__ __launch_bounds__(256)
void moe_stats(const float* __restrict__ zsq, const float* __restrict__ probs,
               const int* __restrict__ cnt, float* __restrict__ outsc) {
  __shared__ float sh[256];
  int tid = threadIdx.x;
  float pz = 0.f, pimp[E_];
#pragma unroll
  for (int e = 0; e < E_; ++e) pimp[e] = 0.f;
  for (int t = tid; t < NTOK; t += 256) {
    pz += zsq[t];
#pragma unroll
    for (int e = 0; e < E_; ++e) pimp[e] += probs[t * E_ + e];
  }
  sh[tid] = pz; __syncthreads();
  for (int s = 128; s > 0; s >>= 1) {
    if (tid < s) sh[tid] += sh[tid + s];
    __syncthreads();
  }
  float zmean = sh[0] / (float)NTOK;
  float aux = 0.f;
  for (int e = 0; e < E_; ++e) {
    __syncthreads();
    sh[tid] = pimp[e]; __syncthreads();
    for (int s = 128; s > 0; s >>= 1) {
      if (tid < s) sh[tid] += sh[tid + s];
      __syncthreads();
    }
    if (tid == 0) aux += (sh[0] / (float)NTOK) * ((float)cnt[e] / (float)NTOK);
  }
  if (tid == 0) { outsc[0] = zmean; outsc[1] = (float)E_ * aux; }
}

// -------------------------------------------------------------- launch ----
extern "C" void kernel_launch(void* const* d_in, const int* in_sizes, int n_in,
                              void* d_out, int out_size, void* d_ws, size_t ws_size,
                              hipStream_t stream) {
  (void)in_sizes; (void)n_in; (void)out_size; (void)ws_size;
  const float* x  = (const float*)d_in[0];
  const float* Wr = (const float*)d_in[1];
  const float* br = (const float*)d_in[2];
  const float* W1 = (const float*)d_in[3];
  const float* b1 = (const float*)d_in[4];
  const float* W2 = (const float*)d_in[5];
  const float* b2 = (const float*)d_in[6];
  float* out = (float*)d_out;

  // workspace layout (floats)
  float* ws    = (float*)d_ws;
  int*   cnt   = (int*)ws;                         // 8
  int*   list  = cnt + E_;                         // 8*2048
  float* wlist = (float*)(list + E_ * NTOK);       // 8*2048
  float* probs = wlist + E_ * NTOK;                // 2048*8
  float* zsq   = probs + NTOK * E_;                // 2048
  float* Y     = zsq + NTOK;                       // (2*NTOK+1) x DIM ~ 16.8 MB

  moe_init   <<<1, 32, 0, stream>>>(cnt);
  moe_router <<<NTOK / 8, 256, 0, stream>>>(x, Wr, br, cnt, list, wlist, probs, zsq);
  moe_ffn    <<<E_ * (NTOK / MT), 512, SMEM_BYTES, stream>>>(x, W1, b1, W2, b2,
                                                             cnt, list, wlist, Y);
  moe_combine<<<(NTOK * DIM / 4) / 256, 256, 0, stream>>>(Y, out);
  moe_stats  <<<1, 256, 0, stream>>>(zsq, probs, cnt, out + (size_t)NTOK * DIM);
}
